// ConvCapsuleLayer_2499670966846
// MI455X (gfx1250) — matrix operand order repaired and checked by reading, not compiled, not run on hardware
//
#include <hip/hip_runtime.h>

typedef __attribute__((ext_vector_type(16))) __bf16 v16bf;
typedef __attribute__((ext_vector_type(8)))  float  v8f;

#define BATCH 16
#define HGT   32
#define WID   32
#define ICAPS 8
#define DIN   16
#define OCAPS 16
#define CDIM  16
// GEMM: M = ICAPS*BATCH*HGT*WID = 131072, N = OCAPS*CDIM = 256, K = 25*16 = 400
#define NTAPS   25
#define KCHUNKS 13          // ceil(25/2) tap-pairs -> K=32 per WMMA
#define MTILES  (ICAPS*BATCH*HGT*WID/16)   // 8192
#define NTILES  (256/16)                   // 16, one tile per output capsule

union AB32 { v16bf v; uint4 u[2]; };

// ---------------- stage 1: x [B,H,W,I,Din] f32 -> xh [I,B,H,W,Din] bf16 ----------------
__global__ void k_cvt_x(const float* __restrict__ x, __bf16* __restrict__ xh) {
  int id = blockIdx.x * 256 + threadIdx.x;
  if (id >= BATCH * HGT * WID * ICAPS * DIN) return;
  int cin = id & 15;
  int i   = (id >> 4)  & 7;
  int w   = (id >> 7)  & 31;
  int h   = (id >> 12) & 31;
  int b   =  id >> 17;
  size_t dst = ((((size_t)i * BATCH + b) * HGT + h) * WID + w) * DIN + cin;
  xh[dst] = (__bf16)x[id];
}

// ---------------- stage 2: pack W [5,5,16,256] f32 into per-lane WMMA B fragments ------
// Bp layout: ((tn*KCHUNKS + c)*32 + lane) * 16 bf16 (contiguous 32B per lane)
// lanes 0-15 hold K=0..15 (tap 2c, cin 0..15) for column n = tn*16 + lane
// lanes 16-31 hold K=16..31 (tap 2c+1, cin 0..15) for column n = tn*16 + lane-16
__global__ void k_pack_w(const float* __restrict__ W, __bf16* __restrict__ Bp) {
  int t = blockIdx.x * 256 + threadIdx.x;
  if (t >= NTILES * KCHUNKS * 32) return;
  int lane = t & 31;
  int c    = (t >> 5) % KCHUNKS;
  int tn   = t / (32 * KCHUNKS);
  int n    = tn * 16 + (lane & 15);
  int tap  = 2 * c + (lane >> 4);
  __bf16* dst = Bp + (size_t)t * 16;
#pragma unroll
  for (int k = 0; k < 16; ++k) {
    float v = (tap < NTAPS) ? W[(size_t)(tap * 16 + k) * 256 + n] : 0.0f;
    dst[k] = (__bf16)v;
  }
}

// ---------------- stage 3: implicit-GEMM conv via v_wmma_f32_16x16x32_bf16 -------------
// One wave per 16-row M-tile covering the FULL N=256 (16 output capsules).
// All 13 A fragments are loaded ONCE into registers (104 VGPRs), then the rolled
// tn-loop runs 13 chained WMMAs per capsule with a single live v8f accumulator,
// keeping peak pressure ~140 VGPRs (no spills).
__device__ __forceinline__ uint4 load_tap(const __bf16* __restrict__ xh, size_t ibbase,
                                          int h, int w, int tap, int khalf) {
  int ky = tap / 5, kx = tap % 5;
  int hh = h + ky - 2, ww = w + kx - 2;
  uint4 z = {0u, 0u, 0u, 0u};
  if (hh < 0 || hh > 31 || ww < 0 || ww > 31) return z;
  const uint4* p =
      (const uint4*)(xh + ((ibbase + (size_t)hh * WID + ww) * DIN + khalf * 8));
  return *p;
}

__global__ void k_conv_wmma(const __bf16* __restrict__ xh, const __bf16* __restrict__ Bp,
                            float* __restrict__ votes) {
  int wave  = (blockIdx.x * blockDim.x + threadIdx.x) >> 5;  // 8192 waves, one per M-tile
  int lane  = threadIdx.x & 31;
  int mbase = wave << 4;
  int i  =  mbase >> 14;                 // input capsule
  int b  = (mbase >> 10) & 15;
  int h  = (mbase >> 5)  & 31;
  int w0 =  mbase & 31;                  // 0 or 16
  int row   = lane & 15;
  int khalf = lane >> 4;
  int w = w0 + row;                      // A-fragment row M = lane%16
  size_t ibbase = (size_t)(i * BATCH + b) * (HGT * WID);

  // Preload all K-chunk A fragments (26 bounds-checked b128 loads, 104 VGPRs)
  AB32 A[KCHUNKS];
#pragma unroll
  for (int c = 0; c < KCHUNKS; ++c) {
    int t0 = 2 * c + khalf;
    if (t0 < NTAPS) { A[c].u[0] = load_tap(xh, ibbase, h, w, t0, 0); A[c].u[1] = load_tap(xh, ibbase, h, w, t0, 1); }
    else            { A[c].u[0] = (uint4){0u, 0u, 0u, 0u}; A[c].u[1] = (uint4){0u, 0u, 0u, 0u}; }
  }

  const uint4* bp_lane = (const uint4*)(Bp + (size_t)lane * 16);
  int d  = lane & 15;
  int hi = lane >> 4;
  size_t site = (size_t)(b * ICAPS + i) * (HGT * WID) + (size_t)h * WID;
  float* vbase = votes + (site + (size_t)w0) * 256 + d;

#pragma unroll 1
  for (int tn = 0; tn < NTILES; ++tn) {
    v8f acc = {};
#pragma unroll
    for (int c = 0; c < KCHUNKS; ++c) {
      AB32 Bv;
      // Fragment for (tn, c): ((tn*KCHUNKS + c)*32 + lane)*16 bf16 -> 64 uint4 stride
      const uint4* bp = bp_lane + (size_t)(tn * KCHUNKS + c) * 64;
      Bv.u[0] = bp[0];
      Bv.u[1] = bp[1];
      acc = __builtin_amdgcn_wmma_f32_16x16x32_bf16(
          false, A[c].v, false, Bv.v, (short)0, acc, false, false);
    }
    // D layout: lane L, vgpr r -> row M = r + 8*(L/16), col = L%16
    // votes[b][i][h][w0+M][o=tn][d=col]
#pragma unroll
    for (int r = 0; r < 8; ++r) {
      vbase[(size_t)(r + 8 * hi) * 256 + tn * 16] = acc[r];
    }
  }
}

// ---------------- stage 4: dynamic routing, one wave per (b,h,w) site ------------------
__global__ void k_routing(const float* __restrict__ votes, const float* __restrict__ bias,
                          float* __restrict__ out) {
  int site = (blockIdx.x * blockDim.x + threadIdx.x) >> 5;  // 16384 sites
  int lane = threadIdx.x & 31;
  int o  = lane & 15;
  int dh = lane >> 4;          // d = dh*8 + j
  int b  = site >> 10;
  int sp = site & 1023;

  float v[ICAPS][8];
#pragma unroll
  for (int i = 0; i < ICAPS; ++i) {
    const float4* p = (const float4*)(votes +
        (((size_t)(b * ICAPS + i) * (HGT * WID) + sp) * 256 + o * 16 + dh * 8));
    float4 lo = p[0], hi = p[1];
    v[i][0] = lo.x; v[i][1] = lo.y; v[i][2] = lo.z; v[i][3] = lo.w;
    v[i][4] = hi.x; v[i][5] = hi.y; v[i][6] = hi.z; v[i][7] = hi.w;
  }
  float bs[8];
#pragma unroll
  for (int j = 0; j < 8; ++j) bs[j] = bias[o * 16 + dh * 8 + j];

  float logit[ICAPS];
#pragma unroll
  for (int i = 0; i < ICAPS; ++i) logit[i] = 0.0f;
  float act[8];

  for (int it = 0; it < 3; ++it) {
    // softmax over o: reduce across the 16 lanes sharing this d-half
    float route[ICAPS];
#pragma unroll
    for (int i = 0; i < ICAPS; ++i) {
      float m = logit[i];
      for (int msk = 1; msk < 16; msk <<= 1) m = fmaxf(m, __shfl_xor(m, msk, 32));
      float e = __expf(logit[i] - m);
      float s = e;
      for (int msk = 1; msk < 16; msk <<= 1) s += __shfl_xor(s, msk, 32);
      route[i] = e / s;
    }
    float pre[8];
#pragma unroll
    for (int j = 0; j < 8; ++j) {
      float s = bs[j];
#pragma unroll
      for (int i = 0; i < ICAPS; ++i) s += route[i] * v[i][j];
      pre[j] = s;
    }
    float n2 = 0.0f;
#pragma unroll
    for (int j = 0; j < 8; ++j) n2 += pre[j] * pre[j];
    n2 += __shfl_xor(n2, 16, 32);              // combine the two d-halves
    float scale = __fsqrt_rn(n2) / (1.0f + n2);
#pragma unroll
    for (int j = 0; j < 8; ++j) act[j] = pre[j] * scale;
    if (it < 2) {
#pragma unroll
      for (int i = 0; i < ICAPS; ++i) {
        float a = 0.0f;
#pragma unroll
        for (int j = 0; j < 8; ++j) a += v[i][j] * act[j];
        a += __shfl_xor(a, 16, 32);
        logit[i] += a;
      }
    }
  }
  // out [B,H,W,O,D]
  float4* q = (float4*)(out + (((size_t)b * (HGT * WID) + sp) * 256 + o * 16 + dh * 8));
  q[0] = make_float4(act[0], act[1], act[2], act[3]);
  q[1] = make_float4(act[4], act[5], act[6], act[7]);
}

extern "C" void kernel_launch(void* const* d_in, const int* in_sizes, int n_in,
                              void* d_out, int out_size, void* d_ws, size_t ws_size,
                              hipStream_t stream) {
  const float* x = (const float*)d_in[0];
  const float* W = (const float*)d_in[1];
  const float* b = (const float*)d_in[2];
  float* out = (float*)d_out;
  char* ws = (char*)d_ws;

  // ws layout: [0,4MB) xh bf16; [4MB,8MB) packed B fragments; [8MB, 8MB+134MB) votes f32
  __bf16* xh    = (__bf16*)ws;
  __bf16* Bp    = (__bf16*)(ws + ((size_t)4 << 20));
  float*  votes = (float*)(ws + ((size_t)8 << 20));

  const int NX = BATCH * HGT * WID * ICAPS * DIN;         // 2,097,152
  k_cvt_x<<<(NX + 255) / 256, 256, 0, stream>>>(x, xh);

  const int NP = NTILES * KCHUNKS * 32;                    // 6656
  k_pack_w<<<(NP + 255) / 256, 256, 0, stream>>>(W, Bp);

  // 8192 M-tiles, one wave each covering full N=256; 8 waves/block
  k_conv_wmma<<<1024, 256, 0, stream>>>(xh, Bp, votes);

  // 16384 sites, one wave each, 8 waves/block
  k_routing<<<2048, 256, 0, stream>>>(votes, b, out);
}
